// R_GNN_Policy_70927089926437
// MI455X (gfx1250) — matrix operand order repaired and checked by reading, not compiled, not run on hardware
//
#include <hip/hip_runtime.h>
#include <cstdint>
#include <cstddef>

typedef __attribute__((ext_vector_type(16))) _Float16 v16h;
typedef __attribute__((ext_vector_type(8)))  _Float16 v8h;
typedef __attribute__((ext_vector_type(8)))  float    v8f;

union V16 { v16h v; v8h h8[2]; _Float16 e[16]; };

#define WMMA16(A,B,C) __builtin_amdgcn_wmma_f32_16x16x32_f16(false,(A),false,(B),(short)0,(C),false,false)

#define BN    4096   // B*N rows
#define NNODE 1024
#define HDIM  128
#define NHEAD 4

__device__ __forceinline__ float sigmoidf_(float x) { return 1.f / (1.f + __expf(-x)); }

// ---------------- small utility kernels ----------------

__global__ void k_cast_f16(const float* __restrict__ src, _Float16* __restrict__ dst, int n) {
    int i = blockIdx.x * blockDim.x + threadIdx.x;
    if (i < n) dst[i] = (_Float16)src[i];
}

// dst[n*K + k] = w[k*N + n]  (weight (K,N) row-major -> (N,K) f16)
__global__ void k_w_colmajor_f16(const float* __restrict__ w, _Float16* __restrict__ dst, int K, int N) {
    int i = blockIdx.x * blockDim.x + threadIdx.x;
    if (i >= K * N) return;
    int k = i / N, n = i % N;
    dst[(size_t)n * K + k] = (_Float16)w[i];
}

// pack adjacency rows into bit-words: adjp[row*32 + w] bits = adj[row][w*32..w*32+31] != 0
__global__ void k_pack_adj(const int* __restrict__ adj, unsigned* __restrict__ adjp) {
    int tid = blockIdx.x * blockDim.x + threadIdx.x;   // BN*32
    if (tid >= BN * 32) return;
    int w = tid & 31, row = tid >> 5;
    const int* a = adj + (size_t)row * NNODE + w * 32;
    unsigned v = 0;
    #pragma unroll
    for (int b = 0; b < 32; ++b) v |= (a[b] != 0 ? 1u : 0u) << b;
    adjp[tid] = v;
}

// ---------------- templated WMMA GEMM (32 rows x NF*16 cols per wave) ----------------
// C(M×N) = act( A(M×K,f16,row-major) * Wc(N×K,f16) + bias(+bias2) )
// OMODE: 0 = f32 row-major, 1 = f16 row-major, 2 = f16 transposed (col*M + row)
template<int K, int NF, int ACT, int OMODE, bool HASB1, bool HASB2>
__global__ void k_gemm_t(const _Float16* __restrict__ A, const _Float16* __restrict__ Wc,
                         const float* __restrict__ bias, const float* __restrict__ bias2,
                         void* __restrict__ outp, int M, int N) {
    const int tileW = NF * 16;
    int tilesN = N / tileW;
    int task = blockIdx.x * (blockDim.x >> 5) + (threadIdx.x >> 5);
    int total = (M >> 5) * tilesN;             // 32 rows per wave
    if (task >= total) return;                 // wave-uniform
    int mt = task / tilesN, nt = task % tilesN;
    int m0 = mt << 5, n0 = nt * tileW;
    int lane = threadIdx.x & 31;
    int kh = lane >> 4, rl = lane & 15;

    // single base pointer per stream; all tile strides become immediate offsets
    const _Float16* arow = A  + (size_t)(m0 + rl) * K + kh * 8;   // +16*K for 2nd row group
    const _Float16* bcol = Wc + (size_t)(n0 + rl) * K + kh * 8;   // +t*16*K per col fragment

    v8f acc0[NF] = {};
    v8f acc1[NF] = {};
    for (int k0 = 0; k0 < K; k0 += 32) {
        V16 a0, a1;
        a0.h8[0] = *(const v8h*)(arow + k0);
        a0.h8[1] = *(const v8h*)(arow + k0 + 16);
        a1.h8[0] = *(const v8h*)(arow + k0 + 16 * K);
        a1.h8[1] = *(const v8h*)(arow + k0 + 16 * K + 16);
        #pragma unroll
        for (int t = 0; t < NF; ++t) {
            V16 bb;
            bb.h8[0] = *(const v8h*)(bcol + k0 + t * 16 * K);
            bb.h8[1] = *(const v8h*)(bcol + k0 + t * 16 * K + 16);
            acc0[t] = WMMA16(a0.v, bb.v, acc0[t]);
            acc1[t] = WMMA16(a1.v, bb.v, acc1[t]);
        }
    }
    #pragma unroll
    for (int g = 0; g < 2; ++g) {
        #pragma unroll
        for (int t = 0; t < NF; ++t) {
            int col = n0 + t * 16 + rl;
            float bv = 0.f;
            if (HASB1) bv += bias[col];
            if (HASB2) bv += bias2[col];
            #pragma unroll
            for (int v = 0; v < 8; ++v) {
                int row = m0 + g * 16 + v + 8 * kh;
                float val = (g == 0 ? acc0[t][v] : acc1[t][v]) + bv;
                if (ACT == 1) val = val > 0.f ? val : 0.f;
                if (OMODE == 0)      ((float*)outp)[(size_t)row * N + col] = val;
                else if (OMODE == 1) ((_Float16*)outp)[(size_t)row * N + col] = (_Float16)val;
                else                 ((_Float16*)outp)[(size_t)col * M + row] = (_Float16)val;
            }
        }
    }
}

// ---------------- GAT attention pieces ----------------

// s1/s2 per (row, head) from hpT (feature-major f16: hpT[(head*128+f)*BN + row])
__global__ void k_scores(const _Float16* __restrict__ hpT, const float* __restrict__ avec,
                         float* __restrict__ s1, float* __restrict__ s2) {
    int tid = blockIdx.x * blockDim.x + threadIdx.x;   // BN*NHEAD
    if (tid >= BN * NHEAD) return;
    int row = tid >> 2, head = tid & 3;
    const _Float16* base = hpT + (size_t)head * HDIM * BN + row;
    float a1 = 0.f, a2 = 0.f;
    for (int f = 0; f < HDIM; ++f) {
        float hv = (float)base[(size_t)f * BN];
        a1 += hv * avec[f];
        a2 += hv * avec[HDIM + f];
    }
    s1[tid] = a1; s2[tid] = a2;
}

// online softmax stats over masked neighbors: m = rowmax, rl = 1/rowsum(exp(e-m))
__global__ void k_stats(const float* __restrict__ s1, const float* __restrict__ s2,
                        const unsigned* __restrict__ adjp,
                        float* __restrict__ mOut, float* __restrict__ rlOut) {
    int tid = blockIdx.x * blockDim.x + threadIdx.x;   // BN*NHEAD
    if (tid >= BN * NHEAD) return;
    int row = tid >> 2, head = tid & 3;
    int b = row >> 10;
    const unsigned* arow = adjp + (size_t)row * 32;
    const float* s2b = s2 + (size_t)(b << 10) * NHEAD + head;
    float s1v = s1[tid];
    float m = -__builtin_inff(), l = 0.f;
    for (int w = 0; w < 32; ++w) {
        unsigned word = arow[w];
        if (!word) continue;
        for (int bit = 0; bit < 32; ++bit) {
            if ((word >> bit) & 1u) {
                float e = s1v + s2b[(w * 32 + bit) * NHEAD];
                e = e > 0.f ? e : 0.2f * e;
                if (e > m) { l = l * __expf(m - e) + 1.f; m = e; }
                else       { l += __expf(e - m); }
            }
        }
    }
    mOut[tid]  = m;
    rlOut[tid] = (l > 0.f) ? 1.f / l : 0.f;
}

// one wave per (b, head, 32-row i-tile): out_i = sum_j softmax(e)_ij * hp_j via WMMA.
// Two P fragments share every hp B-fragment -> 16 WMMA per 32-j chunk.
// MODE 0: elu + concat -> outCat f16 (row-major, width 512)
// MODE 1: raw per-head -> outHead f32 [b][head][i][f]
template<int MODE>
__global__ void k_attn(const _Float16* __restrict__ hpT,
                       const float* __restrict__ s1, const float* __restrict__ s2,
                       const float* __restrict__ mIn, const float* __restrict__ rlIn,
                       const unsigned* __restrict__ adjp,
                       _Float16* __restrict__ outCat, float* __restrict__ outHead) {
    int task = blockIdx.x * (blockDim.x >> 5) + (threadIdx.x >> 5);
    if (task >= 4 * NHEAD * (NNODE / 32)) return;      // wave-uniform
    int b = task >> 7, head = (task >> 5) & 3, it = task & 31;
    int i0 = it * 32;
    int lane = threadIdx.x & 31;
    int kh = lane >> 4, rl = lane & 15;

    int gi0 = (b << 10) + i0 + rl;
    int gi1 = gi0 + 16;
    float s1v0 = s1[gi0 * NHEAD + head],  s1v1 = s1[gi1 * NHEAD + head];
    float mv0  = mIn[gi0 * NHEAD + head], mv1  = mIn[gi1 * NHEAD + head];
    float rlv0 = rlIn[gi0 * NHEAD + head], rlv1 = rlIn[gi1 * NHEAD + head];
    const unsigned* arow0 = adjp + (size_t)gi0 * 32;
    const unsigned* arow1 = adjp + (size_t)gi1 * 32;
    const float* s2b = s2 + (size_t)(b << 10) * NHEAD + head;
    // base pointer for hp value fragments; per-nf stride is an immediate (nf*16*BN elems)
    const _Float16* hbrl = hpT + (size_t)head * HDIM * BN + (size_t)rl * BN + (b << 10) + kh * 16;

    v8f acc0[8] = {};
    v8f acc1[8] = {};
    for (int jc = 0; jc < NNODE; jc += 32) {
        unsigned word0 = arow0[jc >> 5];
        unsigned word1 = arow1[jc >> 5];
        V16 p0, p1;
        #pragma unroll
        for (int t = 0; t < 16; ++t) {
            int jl = kh * 16 + t;
            float sv = s2b[(jc + jl) * NHEAD];
            float e0 = s1v0 + sv;  e0 = e0 > 0.f ? e0 : 0.2f * e0;
            float e1 = s1v1 + sv;  e1 = e1 > 0.f ? e1 : 0.2f * e1;
            float pv0 = ((word0 >> jl) & 1u) ? __expf(e0 - mv0) * rlv0 : 0.f;
            float pv1 = ((word1 >> jl) & 1u) ? __expf(e1 - mv1) * rlv1 : 0.f;
            p0.e[t] = (_Float16)pv0;
            p1.e[t] = (_Float16)pv1;
        }
        #pragma unroll
        for (int nf = 0; nf < 8; ++nf) {
            V16 bb;
            bb.h8[0] = *(const v8h*)(hbrl + (size_t)nf * 16 * BN + jc);
            bb.h8[1] = *(const v8h*)(hbrl + (size_t)nf * 16 * BN + jc + 8);
            acc0[nf] = WMMA16(p0.v, bb.v, acc0[nf]);
            acc1[nf] = WMMA16(p1.v, bb.v, acc1[nf]);
        }
    }
    #pragma unroll
    for (int g = 0; g < 2; ++g) {
        #pragma unroll
        for (int nf = 0; nf < 8; ++nf) {
            #pragma unroll
            for (int v = 0; v < 8; ++v) {
                int irow = i0 + g * 16 + v + 8 * kh;
                int gr   = (b << 10) + irow;
                int f    = nf * 16 + rl;
                float val = (g == 0 ? acc0[nf][v] : acc1[nf][v]);
                if (MODE == 0) {
                    val = val > 0.f ? val : expm1f(val);
                    outCat[(size_t)gr * 512 + head * HDIM + f] = (_Float16)val;
                } else {
                    outHead[(((size_t)(b * NHEAD + head) * NNODE) + irow) * HDIM + f] = val;
                }
            }
        }
    }
}

__global__ void k_mean_heads(const float* __restrict__ att2, _Float16* __restrict__ out16) {
    int tid = blockIdx.x * blockDim.x + threadIdx.x;   // BN*HDIM
    if (tid >= BN * HDIM) return;
    int mrow = tid >> 7, f = tid & 127;
    int b = mrow >> 10, i = mrow & 1023;
    float s = 0.f;
    #pragma unroll
    for (int h = 0; h < NHEAD; ++h)
        s += att2[(((size_t)(b * NHEAD + h) * NNODE) + i) * HDIM + f];
    out16[tid] = (_Float16)(0.25f * s);
}

__global__ void k_lstm_elem(const float* __restrict__ gates,
                            float* __restrict__ hOut, float* __restrict__ cOut,
                            _Float16* __restrict__ h16) {
    int tid = blockIdx.x * blockDim.x + threadIdx.x;   // BN*HDIM
    if (tid >= BN * HDIM) return;
    int m = tid >> 7, f = tid & 127;
    const float* g = gates + (size_t)m * 512;
    float iv = g[f], gv = g[256 + f], ov = g[384 + f];
    float c = sigmoidf_(iv) * tanhf(gv);
    float h = sigmoidf_(ov) * tanhf(c);
    cOut[tid] = c;
    hOut[tid] = h;
    h16[tid]  = (_Float16)h;
}

// ---------------- host side ----------------

static inline int gemm_blocks(int M, int N, int tileW) {
    int tasks = (M / 32) * (N / tileW);
    return (tasks + 3) / 4;
}

extern "C" void kernel_launch(void* const* d_in, const int* in_sizes, int n_in,
                              void* d_out, int out_size, void* d_ws, size_t ws_size,
                              hipStream_t stream) {
    (void)in_sizes; (void)n_in; (void)out_size; (void)ws_size;
    const float* nodeF    = (const float*)d_in[0];
    const int*   adj      = (const int*)d_in[1];
    const float* enc_w1   = (const float*)d_in[2];
    const float* enc_b1   = (const float*)d_in[3];
    const float* enc_w2   = (const float*)d_in[4];
    const float* enc_b2   = (const float*)d_in[5];
    const float* gatW[3]  = {(const float*)d_in[6], (const float*)d_in[8], (const float*)d_in[10]};
    const float* gatA[3]  = {(const float*)d_in[7], (const float*)d_in[9], (const float*)d_in[11]};
    const float* lstm_wih = (const float*)d_in[12];
    const float* lstm_bih = (const float*)d_in[14];
    const float* lstm_bhh = (const float*)d_in[15];
    const float* dec_w1   = (const float*)d_in[16];
    const float* dec_b1   = (const float*)d_in[17];
    const float* dec_w2   = (const float*)d_in[18];
    const float* dec_b2   = (const float*)d_in[19];

    char* ws = (char*)d_ws;
    size_t off = 0;
    auto AL = [&](size_t bytes) { size_t r = off; off += (bytes + 255) & ~(size_t)255; return r; };

    _Float16* x16  = (_Float16*)(ws + AL((size_t)BN * 64 * 2));
    _Float16* bufA = (_Float16*)(ws + AL((size_t)BN * 512 * 2));
    _Float16* bufB = (_Float16*)(ws + AL((size_t)BN * 512 * 2));
    _Float16* hpT  = (_Float16*)(ws + AL((size_t)512 * BN * 2));
    float* s1   = (float*)(ws + AL((size_t)BN * NHEAD * 4));
    float* s2   = (float*)(ws + AL((size_t)BN * NHEAD * 4));
    float* mBuf = (float*)(ws + AL((size_t)BN * NHEAD * 4));
    float* rlB  = (float*)(ws + AL((size_t)BN * NHEAD * 4));
    unsigned* adjp = (unsigned*)(ws + AL((size_t)BN * 32 * 4));
    float* att2  = (float*)(ws + AL((size_t)BN * NHEAD * HDIM * 4));
    float* gates = (float*)(ws + AL((size_t)BN * 512 * 4));
    _Float16* wE1 = (_Float16*)(ws + AL((size_t)128 * 64 * 2));
    _Float16* wE2 = (_Float16*)(ws + AL((size_t)128 * 128 * 2));
    _Float16* wG[3];
    wG[0] = (_Float16*)(ws + AL((size_t)512 * 128 * 2));
    wG[1] = (_Float16*)(ws + AL((size_t)512 * 512 * 2));
    wG[2] = (_Float16*)(ws + AL((size_t)512 * 512 * 2));
    _Float16* wL  = (_Float16*)(ws + AL((size_t)512 * 128 * 2));
    _Float16* wD1 = (_Float16*)(ws + AL((size_t)128 * 128 * 2));
    _Float16* wD2 = (_Float16*)(ws + AL((size_t)16 * 128 * 2));

    float* qOut = (float*)d_out;
    float* hNew = qOut + BN * 16;
    float* cNew = hNew + BN * HDIM;

    auto g1 = [](int n) { return (n + 255) / 256; };

    // -------- weight / input conversion --------
    k_cast_f16<<<g1(BN * 64), 256, 0, stream>>>(nodeF, x16, BN * 64);
    k_w_colmajor_f16<<<g1(64 * 128), 256, 0, stream>>>(enc_w1, wE1, 64, 128);
    k_w_colmajor_f16<<<g1(128 * 128), 256, 0, stream>>>(enc_w2, wE2, 128, 128);
    k_w_colmajor_f16<<<g1(128 * 512), 256, 0, stream>>>(gatW[0], wG[0], 128, 512);
    k_w_colmajor_f16<<<g1(512 * 512), 256, 0, stream>>>(gatW[1], wG[1], 512, 512);
    k_w_colmajor_f16<<<g1(512 * 512), 256, 0, stream>>>(gatW[2], wG[2], 512, 512);
    k_cast_f16<<<g1(512 * 128), 256, 0, stream>>>(lstm_wih, wL, 512 * 128); // already (N=512,K=128)
    k_w_colmajor_f16<<<g1(128 * 128), 256, 0, stream>>>(dec_w1, wD1, 128, 128);
    k_w_colmajor_f16<<<g1(128 * 16), 256, 0, stream>>>(dec_w2, wD2, 128, 16);
    k_pack_adj<<<g1(BN * 32), 256, 0, stream>>>(adj, adjp);

    // -------- encoder --------
    k_gemm_t<64, 4, 1, 1, true, false><<<gemm_blocks(BN, 128, 64), 128, 0, stream>>>(
        x16, wE1, enc_b1, nullptr, bufA, BN, 128);
    k_gemm_t<128, 4, 0, 1, true, false><<<gemm_blocks(BN, 128, 64), 128, 0, stream>>>(
        bufA, wE2, enc_b2, nullptr, bufB, BN, 128);

    // -------- GAT layers --------
    _Float16* Ain[3]  = {bufB, bufA, bufB};
    _Float16* Aout[3] = {bufA, bufB, nullptr};
    for (int l = 0; l < 3; ++l) {
        if (l == 0)
            k_gemm_t<128, 4, 0, 2, false, false><<<gemm_blocks(BN, 512, 64), 128, 0, stream>>>(
                Ain[l], wG[l], nullptr, nullptr, hpT, BN, 512);
        else
            k_gemm_t<512, 4, 0, 2, false, false><<<gemm_blocks(BN, 512, 64), 128, 0, stream>>>(
                Ain[l], wG[l], nullptr, nullptr, hpT, BN, 512);
        k_scores<<<g1(BN * NHEAD), 256, 0, stream>>>(hpT, gatA[l], s1, s2);
        k_stats<<<g1(BN * NHEAD), 256, 0, stream>>>(s1, s2, adjp, mBuf, rlB);
        if (l < 2)
            k_attn<0><<<128, 128, 0, stream>>>(hpT, s1, s2, mBuf, rlB, adjp, Aout[l], nullptr);
        else
            k_attn<1><<<128, 128, 0, stream>>>(hpT, s1, s2, mBuf, rlB, adjp, nullptr, att2);
    }
    k_mean_heads<<<g1(BN * HDIM), 256, 0, stream>>>(att2, bufA);

    // -------- LSTM cell --------
    k_gemm_t<128, 4, 0, 0, true, true><<<gemm_blocks(BN, 512, 64), 128, 0, stream>>>(
        bufA, wL, lstm_bih, lstm_bhh, gates, BN, 512);
    k_lstm_elem<<<g1(BN * HDIM), 256, 0, stream>>>(gates, hNew, cNew, bufB);

    // -------- decoder --------
    k_gemm_t<128, 4, 1, 1, true, false><<<gemm_blocks(BN, 128, 64), 128, 0, stream>>>(
        bufB, wD1, dec_b1, nullptr, bufA, BN, 128);
    k_gemm_t<128, 1, 0, 0, true, false><<<gemm_blocks(BN, 16, 16), 128, 0, stream>>>(
        bufA, wD2, dec_b2, nullptr, qOut, BN, 16);
}
